// Generator_18614388260997
// MI455X (gfx1250) — matrix operand order repaired and checked by reading, not compiled, not run on hardware
//
#include <hip/hip_runtime.h>
#include <hip/hip_bf16.h>

// ---------------------------------------------------------------------------
// CDNA5 (gfx1250) generator forward. All convs are implicit-GEMM on
// v_wmma_f32_16x16x32_f16. Each wave computes a 32x64 output slab (2 M-tiles
// x 4 N-tiles): per 32-deep K step, two A fragments (packed fp16 weights) and
// four implicitly-gathered B fragments feed 8 WMMAs, so every im2col gather
// element is reused by 2 matrix ops and every weight fragment by 4.
// ---------------------------------------------------------------------------

typedef __attribute__((ext_vector_type(16))) _Float16 v16h;
typedef __attribute__((ext_vector_type(8)))  _Float16 v8h;
typedef __attribute__((ext_vector_type(8)))  float    v8f;

#define EPSN 1e-5f

__device__ __forceinline__ float sigm(float x) { return 1.f / (1.f + __expf(-x)); }

// ---------------------------------------------------------------------------
// Weight packing: f32 [Cout][Cin*KH*KW] row-major -> f16 [Mp][Kp], zero padded
// ---------------------------------------------------------------------------
__global__ void pack_w_k(const float* __restrict__ w, _Float16* __restrict__ wp,
                         int Cout, int K, int Kp, int Mp) {
    int idx = blockIdx.x * 256 + threadIdx.x;
    int total = Mp * Kp;
    if (idx >= total) return;
    int m = idx / Kp, k = idx - m * Kp;
    float v = (m < Cout && k < K) ? w[(long)m * K + k] : 0.f;
    wp[idx] = (_Float16)v;
}

// ---------------------------------------------------------------------------
// Implicit-GEMM convolution via WMMA. 8 waves / block; wave computes 32x64.
//   act: fp16 [Cin][Bn*H*W];  out: f32 [Cout][Bn*OH*OW]
// A layout (16-bit, 16x32): lane holds row m = lane%16, K elems
//   {half*16+e}.  B: lane holds column p, K elems {half*16+e}.
// ---------------------------------------------------------------------------
__global__ __launch_bounds__(256)
void conv_wmma_k(const _Float16* __restrict__ wp, const _Float16* __restrict__ act,
                 const float* __restrict__ bias, float* __restrict__ out,
                 int Cin, int H, int W, int Bn,
                 int KH, int KW, int SH, int SW, int PH, int PW,
                 int Cout, int OH, int OW, int Kp) {
    const int lane = threadIdx.x & 31;
    const int wave = threadIdx.x >> 5;
    const int half = lane >> 4;
    const int lid  = lane & 15;
    const int P  = Bn * OH * OW;
    const int m0 = blockIdx.y * 32;
    const int n0 = (blockIdx.x * 8 + wave) * 64;

    const int KRS = KH * KW;
    const int K   = Cin * KRS;
    const int HW  = H * W;
    const long planeStride = (long)Bn * HW;

    // Per-tile pixel descriptors (4 N-tiles per wave)
    int ih0[4], iw0[4];
    const _Float16* actb[4];
#pragma unroll
    for (int t = 0; t < 4; ++t) {
        int p = n0 + t * 16 + lid;
        if (p >= P) p = P - 1;                  // clamp; stores are guarded
        int b   = p / (OH * OW);
        int rem = p - b * (OH * OW);
        int oh  = rem / OW;
        int ow  = rem - oh * OW;
        ih0[t] = oh * SH - PH;
        iw0[t] = ow * SW - PW;
        actb[t] = act + (long)b * HW;
    }

    const _Float16* wrow0 = wp + (long)(m0 + lid) * Kp;
    const _Float16* wrow1 = wp + (long)(m0 + 16 + lid) * Kp;

    v8f acc[2][4] = {{v8f{}, v8f{}, v8f{}, v8f{}},
                     {v8f{}, v8f{}, v8f{}, v8f{}}};

    for (int k0 = 0; k0 < Kp; k0 += 32) {
        // ---- A fragments: 16B vector loads (contiguous packed weights) ----
        __builtin_prefetch(wrow0 + k0 + 64, 0, 0);
        __builtin_prefetch(wrow1 + k0 + 64, 0, 0);
        v16h af0, af1;
        {
            const v8h a00 = *(const v8h*)(wrow0 + k0 + half * 16);
            const v8h a01 = *(const v8h*)(wrow0 + k0 + half * 16 + 8);
            const v8h a10 = *(const v8h*)(wrow1 + k0 + half * 16);
            const v8h a11 = *(const v8h*)(wrow1 + k0 + half * 16 + 8);
#pragma unroll
            for (int e = 0; e < 8; ++e) {
                af0[e] = a00[e]; af0[e + 8] = a01[e];
                af1[e] = a10[e]; af1[e + 8] = a11[e];
            }
        }
        // ---- decompose kb once, then walk (c,r,s) incrementally ----
        const int kb = k0 + half * 16;
        int c  = kb / KRS;
        int rs = kb - c * KRS;
        int r  = rs / KW;
        int s  = rs - r * KW;

        v16h bf[4];
#pragma unroll
        for (int e = 0; e < 16; ++e) {
            const bool kok = (kb + e) < K;
            const long cbase = (long)c * planeStride;
#pragma unroll
            for (int t = 0; t < 4; ++t) {
                int ih = ih0[t] + r, iw = iw0[t] + s;
                _Float16 v = (_Float16)0.f;
                if (kok && ih >= 0 && ih < H && iw >= 0 && iw < W)
                    v = actb[t][cbase + (long)ih * W + iw];
                bf[t][e] = v;
            }
            if (++s == KW) { s = 0; if (++r == KH) { r = 0; ++c; } }
        }
        acc[0][0] = __builtin_amdgcn_wmma_f32_16x16x32_f16(false, af0, false, bf[0], (short)0, acc[0][0], false, false);
        acc[1][0] = __builtin_amdgcn_wmma_f32_16x16x32_f16(false, af1, false, bf[0], (short)0, acc[1][0], false, false);
        acc[0][1] = __builtin_amdgcn_wmma_f32_16x16x32_f16(false, af0, false, bf[1], (short)0, acc[0][1], false, false);
        acc[1][1] = __builtin_amdgcn_wmma_f32_16x16x32_f16(false, af1, false, bf[1], (short)0, acc[1][1], false, false);
        acc[0][2] = __builtin_amdgcn_wmma_f32_16x16x32_f16(false, af0, false, bf[2], (short)0, acc[0][2], false, false);
        acc[1][2] = __builtin_amdgcn_wmma_f32_16x16x32_f16(false, af1, false, bf[2], (short)0, acc[1][2], false, false);
        acc[0][3] = __builtin_amdgcn_wmma_f32_16x16x32_f16(false, af0, false, bf[3], (short)0, acc[0][3], false, false);
        acc[1][3] = __builtin_amdgcn_wmma_f32_16x16x32_f16(false, af1, false, bf[3], (short)0, acc[1][3], false, false);
    }

#pragma unroll
    for (int mt = 0; mt < 2; ++mt) {
#pragma unroll
        for (int v = 0; v < 8; ++v) {
            int m = m0 + mt * 16 + half * 8 + v;
            if (m >= Cout) continue;
            float bv = bias ? bias[m] : 0.f;
            float* orow = out + (long)m * P;
#pragma unroll
            for (int t = 0; t < 4; ++t) {
                int n = n0 + t * 16 + lid;
                if (n < P) orow[n] = acc[mt][t][v] + bv;
            }
        }
    }
}

// ---------------------------------------------------------------------------
// Elementwise kernels
// ---------------------------------------------------------------------------
__global__ void build_xs_k(const float* __restrict__ x, const float* __restrict__ m,
                           _Float16* __restrict__ xs, int P) {
    int i = blockIdx.x * 256 + threadIdx.x;
    if (i >= P) return;
    xs[i]     = (_Float16)(x[i] * m[i]);
    xs[P + i] = (_Float16)(m[i]);
}

__global__ void cvt16_k(const float* __restrict__ a, _Float16* __restrict__ o, int n) {
    int i = blockIdx.x * 256 + threadIdx.x;
    if (i < n) o[i] = (_Float16)a[i];
}

__global__ void glu_k(const float* __restrict__ a, const float* __restrict__ g,
                      float* __restrict__ o, int n) {
    int i = blockIdx.x * 256 + threadIdx.x;
    if (i < n) o[i] = a[i] * sigm(g[i]);
}

// GLU fused with fp16 conversion (when only next conv consumes the result)
__global__ void glu16_k(const float* __restrict__ a, const float* __restrict__ g,
                        _Float16* __restrict__ o, int n) {
    int i = blockIdx.x * 256 + threadIdx.x;
    if (i < n) o[i] = (_Float16)(a[i] * sigm(g[i]));
}

// self-GLU fused with fp16 conversion
__global__ void selfglu16_k(const float* __restrict__ x, _Float16* __restrict__ o, int n) {
    int i = blockIdx.x * 256 + threadIdx.x;
    if (i < n) { float v = x[i]; o[i] = (_Float16)(v * sigm(v)); }
}

__global__ void add_k(float* __restrict__ a, const float* __restrict__ b, int n) {
    int i = blockIdx.x * 256 + threadIdx.x;
    if (i < n) a[i] += b[i];
}

// ---------------------------------------------------------------------------
// Instance norm: per (c,b) over S spatial elems, in-place, affine g/be
// ---------------------------------------------------------------------------
__global__ __launch_bounds__(256)
void instnorm_k(float* __restrict__ x, const float* __restrict__ g,
                const float* __restrict__ be, int Bn, int S) {
    int cb = blockIdx.x;
    int c  = cb / Bn;
    int b  = cb - c * Bn;
    float* row = x + ((long)c * Bn + b) * S;
    __shared__ float s1[256], s2[256];
    float a = 0.f, q = 0.f;
    for (int i = threadIdx.x; i < S; i += 256) { float v = row[i]; a += v; q += v * v; }
    s1[threadIdx.x] = a; s2[threadIdx.x] = q; __syncthreads();
    for (int o = 128; o > 0; o >>= 1) {
        if (threadIdx.x < o) { s1[threadIdx.x] += s1[threadIdx.x + o]; s2[threadIdx.x] += s2[threadIdx.x + o]; }
        __syncthreads();
    }
    float mean = s1[0] / (float)S;
    float var  = s2[0] / (float)S - mean * mean;
    float sc = rsqrtf(var + EPSN) * g[c];
    float sh = be[c];
    for (int i = threadIdx.x; i < S; i += 256) row[i] = (row[i] - mean) * sc + sh;
}

// ---------------------------------------------------------------------------
// RN_B: masked dual BN over (batch,length). mask is (B,80,512); m1(c,b,l) =
// (c<80) ? mask[b][c][l] : 0 (L=128 <= 512 so all l valid).
// ---------------------------------------------------------------------------
__global__ __launch_bounds__(256)
void rnb_stats_k(const float* __restrict__ x, const float* __restrict__ mask,
                 float* __restrict__ st, int Bn, int L) {
    int c = blockIdx.x;
    const float* row = x + (long)c * Bn * L;
    float sf = 0.f, qf = 0.f, sb = 0.f, qb = 0.f;
    for (int i = threadIdx.x; i < Bn * L; i += 256) {
        int b = i / L, l = i - b * L;
        float m = (c < 80) ? mask[((long)b * 80 + c) * 512 + l] : 0.f;
        float v = row[i];
        float xm = v * m, xb = v * (1.f - m);
        sf += xm; qf += xm * xm; sb += xb; qb += xb * xb;
    }
    __shared__ float s[4][256];
    s[0][threadIdx.x] = sf; s[1][threadIdx.x] = qf;
    s[2][threadIdx.x] = sb; s[3][threadIdx.x] = qb;
    __syncthreads();
    for (int o = 128; o > 0; o >>= 1) {
        if (threadIdx.x < o)
            for (int j = 0; j < 4; ++j) s[j][threadIdx.x] += s[j][threadIdx.x + o];
        __syncthreads();
    }
    if (threadIdx.x == 0) {
        float N = (float)(Bn * L);
        float mf = s[0][0] / N, mb = s[2][0] / N;
        st[c * 4 + 0] = mf;
        st[c * 4 + 1] = s[1][0] / N - mf * mf;
        st[c * 4 + 2] = mb;
        st[c * 4 + 3] = s[3][0] / N - mb * mb;
    }
}

__global__ void rnb_apply_k(const float* __restrict__ x, const float* __restrict__ mask,
                            const float* __restrict__ st,
                            const float* __restrict__ fg, const float* __restrict__ fb,
                            const float* __restrict__ bg, const float* __restrict__ bb,
                            float* __restrict__ o, int C, int Bn, int L) {
    int i = blockIdx.x * 256 + threadIdx.x;
    int total = C * Bn * L;
    if (i >= total) return;
    int c = i / (Bn * L);
    int t = i - c * (Bn * L);
    int b = t / L, l = t - b * L;
    float m  = (c < 80) ? mask[((long)b * 80 + c) * 512 + l] : 0.f;
    float v  = x[i];
    float mf = st[c * 4 + 0], vf = st[c * 4 + 1];
    float mb = st[c * 4 + 2], vb = st[c * 4 + 3];
    float rn = (v * m - mf) * rsqrtf(vf + EPSN) + (v * (1.f - m) - mb) * rsqrtf(vb + EPSN);
    o[i] = rn * m * (1.f + fg[c]) + fb[c] + rn * (1.f - m) * (1.f + bg[c]) + bb[c];
}

// ---------------------------------------------------------------------------
// Layout shuffles
// ---------------------------------------------------------------------------
// (B,C,R,L) [C][Bn*R*L] -> (B,C*R,L) [C*R][Bn*L], fused f32->f16
__global__ void reshape_2to1_h16_k(const float* __restrict__ in, _Float16* __restrict__ out,
                                   int C, int Bn, int R, int L) {
    int i = blockIdx.x * 256 + threadIdx.x;
    int total = C * R * Bn * L;
    if (i >= total) return;
    int f = i / (Bn * L);
    int t = i - f * (Bn * L);
    int b = t / L, l = t - b * L;
    int c = f / R, r = f - c * R;
    out[i] = (_Float16)in[((long)c * Bn + b) * R * L + r * L + l];
}

// (B,C*R,L) [C*R][Bn*L] -> (B,C,R,L) [C][Bn*R*L], fused f32->f16
__global__ void reshape_1to2_h16_k(const float* __restrict__ in, _Float16* __restrict__ out,
                                   int C, int Bn, int R, int L) {
    int i = blockIdx.x * 256 + threadIdx.x;
    int total = C * Bn * R * L;
    if (i >= total) return;
    int c = i / (Bn * R * L);
    int t = i - c * (Bn * R * L);
    int b = t / (R * L);
    int u = t - b * (R * L);
    int r = u / L, l = u - r * L;
    out[i] = (_Float16)in[((long)(c * R + r) * Bn + b) * L + l];
}

// pixel shuffle r=2: in [4*Co][Bn*H*W] -> out [Co][Bn*2H*2W]
__global__ void pshuf_k(const float* __restrict__ in, float* __restrict__ out,
                        int Co, int Bn, int H, int W) {
    int i = blockIdx.x * 256 + threadIdx.x;
    int total = Co * Bn * 4 * H * W;
    if (i >= total) return;
    int cp = i / (Bn * 4 * H * W);
    int t  = i - cp * (Bn * 4 * H * W);
    int b  = t / (4 * H * W);
    int u  = t - b * (4 * H * W);
    int hp = u / (2 * W);
    int wp = u - hp * (2 * W);
    int ci = cp * 4 + (hp & 1) * 2 + (wp & 1);
    out[i] = in[((long)ci * Bn + b) * H * W + (hp >> 1) * W + (wp >> 1)];
}

// ---------------------------------------------------------------------------
// Host side
// ---------------------------------------------------------------------------
enum {
    IN_X = 0, IN_MASK,
    P_CONV1_W, P_CONV1_B, P_CONV1G_W, P_CONV1G_B,
    P_DS1_W, P_DS1_B, P_DS1_G, P_DS1_BE, P_DS1_GW, P_DS1_GB, P_DS1_GG, P_DS1_GBE,
    P_DS2_W, P_DS2_B, P_DS2_G, P_DS2_BE, P_DS2_GW, P_DS2_GB, P_DS2_GG, P_DS2_GBE,
    P_C21_W, P_C21_B, P_RN_FG, P_RN_FB, P_RN_BG, P_RN_BB,
    P_RES_W1, P_RES_B1, P_RES_G1, P_RES_BE1,
    P_RES_WG, P_RES_BG, P_RES_GG, P_RES_BEG,
    P_RES_W2, P_RES_B2, P_RES_G2, P_RES_BE2,
    P_C12_W, P_C12_B, P_RN1_FG, P_RN1_FB, P_RN1_BG, P_RN1_BB,
    P_UP1_W, P_UP1_B, P_UP1_G, P_UP1_BE,
    P_UP2_W, P_UP2_B, P_UP2_G, P_UP2_BE,
    P_LAST_W, P_LAST_B
};

static inline int ceildiv(int a, int b) { return (a + b - 1) / b; }

static void run_conv(const float* w, const float* bias, const _Float16* act16,
                     float* out, _Float16* wpack,
                     int Cin, int H, int W, int Bn, int KH, int KW,
                     int SH, int SW, int PH, int PW,
                     int Cout, int OH, int OW, hipStream_t s) {
    int K  = Cin * KH * KW;
    int Kp = (K + 31) & ~31;
    int Mp = (Cout + 31) & ~31;     // 2 M-tiles per wave
    int tot = Mp * Kp;
    pack_w_k<<<ceildiv(tot, 256), 256, 0, s>>>(w, wpack, Cout, K, Kp, Mp);
    int P = Bn * OH * OW;
    dim3 g(ceildiv(P, 512), Mp / 32);   // 8 waves/block, 64 px x 32 rows/wave
    conv_wmma_k<<<g, 256, 0, s>>>(wpack, act16, bias, out,
                                  Cin, H, W, Bn, KH, KW, SH, SW, PH, PW,
                                  Cout, OH, OW, Kp);
}

extern "C" void kernel_launch(void* const* d_in, const int* in_sizes, int n_in,
                              void* d_out, int out_size, void* d_ws, size_t ws_size,
                              hipStream_t stream) {
    (void)in_sizes; (void)n_in; (void)out_size; (void)ws_size;
    const int Bn = 8;
    const int P0 = Bn * 80 * 512;   // 327680

    const float* x    = (const float*)d_in[IN_X];
    const float* mask = (const float*)d_in[IN_MASK];
    #define PAR(i) ((const float*)d_in[i])

    // workspace layout
    char* base = (char*)d_ws;
    const size_t SZ_W16 = 16ull << 20;
    const size_t SZ_H16 = 88ull << 20;
    const size_t SZ_F   = 168ull << 20;
    _Float16* w16 = (_Float16*)(base);
    _Float16* h16 = (_Float16*)(base + SZ_W16);
    float* fa = (float*)(base + SZ_W16 + SZ_H16);
    float* fb = (float*)(base + SZ_W16 + SZ_H16 + SZ_F);
    float* fc = (float*)(base + SZ_W16 + SZ_H16 + 2 * SZ_F);
    float* st = (float*)(base + SZ_W16 + SZ_H16 + 3 * SZ_F);

    #define EW(n) ceildiv((n), 256), 256, 0, stream

    // ---- stem: xs = stack(x*mask, mask); conv1 / conv1g 5x15; GLU ----
    build_xs_k<<<EW(P0)>>>(x, mask, h16, P0);
    run_conv(PAR(P_CONV1_W),  PAR(P_CONV1_B),  h16, fa, w16,
             2, 80, 512, Bn, 5, 15, 1, 1, 2, 7, 128, 80, 512, stream);
    run_conv(PAR(P_CONV1G_W), PAR(P_CONV1G_B), h16, fb, w16,
             2, 80, 512, Bn, 5, 15, 1, 1, 2, 7, 128, 80, 512, stream);
    glu16_k<<<EW(128 * P0)>>>(fa, fb, h16, 128 * P0);

    // ---- ds1: 128->256, 5x5 s2, IN, GLU ----
    const int P1 = Bn * 40 * 256;   // 81920
    run_conv(PAR(P_DS1_W),  PAR(P_DS1_B),  h16, fa, w16,
             128, 80, 512, Bn, 5, 5, 2, 2, 2, 2, 256, 40, 256, stream);
    run_conv(PAR(P_DS1_GW), PAR(P_DS1_GB), h16, fb, w16,
             128, 80, 512, Bn, 5, 5, 2, 2, 2, 2, 256, 40, 256, stream);
    instnorm_k<<<256 * Bn, 256, 0, stream>>>(fa, PAR(P_DS1_G),  PAR(P_DS1_BE),  Bn, 40 * 256);
    instnorm_k<<<256 * Bn, 256, 0, stream>>>(fb, PAR(P_DS1_GG), PAR(P_DS1_GBE), Bn, 40 * 256);
    glu16_k<<<EW(256 * P1)>>>(fa, fb, h16, 256 * P1);

    // ---- ds2: 256->256, 5x5 s2, IN, GLU ----
    const int P2 = Bn * 20 * 128;   // 20480
    run_conv(PAR(P_DS2_W),  PAR(P_DS2_B),  h16, fa, w16,
             256, 40, 256, Bn, 5, 5, 2, 2, 2, 2, 256, 20, 128, stream);
    run_conv(PAR(P_DS2_GW), PAR(P_DS2_GB), h16, fb, w16,
             256, 40, 256, Bn, 5, 5, 2, 2, 2, 2, 256, 20, 128, stream);
    instnorm_k<<<256 * Bn, 256, 0, stream>>>(fa, PAR(P_DS2_G),  PAR(P_DS2_BE),  Bn, 20 * 128);
    instnorm_k<<<256 * Bn, 256, 0, stream>>>(fb, PAR(P_DS2_GG), PAR(P_DS2_GBE), Bn, 20 * 128);
    glu_k<<<EW(256 * P2)>>>(fa, fb, fc, 256 * P2);

    // ---- reshape (B,256,20,128) -> (B,5120,128) fp16 ; c21 1x1 ; RN_B ----
    const int L4 = 128, P1d = Bn * L4;  // 1024
    reshape_2to1_h16_k<<<EW(5120 * P1d)>>>(fc, h16, 256, Bn, 20, L4);
    run_conv(PAR(P_C21_W), PAR(P_C21_B), h16, fb, w16,
             5120, 1, L4, Bn, 1, 1, 1, 1, 0, 0, 256, 1, L4, stream);
    rnb_stats_k<<<256, 256, 0, stream>>>(fb, mask, st, Bn, L4);
    rnb_apply_k<<<EW(256 * P1d)>>>(fb, mask, st,
        PAR(P_RN_FG), PAR(P_RN_FB), PAR(P_RN_BG), PAR(P_RN_BB), fc, 256, Bn, L4);

    // ---- 6 residual GLU blocks (h lives in fc) ----
    for (int i = 0; i < 6; ++i) {
        const float* w1 = PAR(P_RES_W1) + (long)i * 512 * 256 * 3;
        const float* wg = PAR(P_RES_WG) + (long)i * 512 * 256 * 3;
        const float* w2 = PAR(P_RES_W2) + (long)i * 256 * 512 * 3;
        cvt16_k<<<EW(256 * P1d)>>>(fc, h16, 256 * P1d);
        run_conv(w1, PAR(P_RES_B1) + i * 512, h16, fa, w16,
                 256, 1, L4, Bn, 1, 3, 1, 1, 0, 1, 512, 1, L4, stream);
        run_conv(wg, PAR(P_RES_BG) + i * 512, h16, fb, w16,
                 256, 1, L4, Bn, 1, 3, 1, 1, 0, 1, 512, 1, L4, stream);
        instnorm_k<<<512 * Bn, 256, 0, stream>>>(fa, PAR(P_RES_G1) + i * 512, PAR(P_RES_BE1) + i * 512, Bn, L4);
        instnorm_k<<<512 * Bn, 256, 0, stream>>>(fb, PAR(P_RES_GG) + i * 512, PAR(P_RES_BEG) + i * 512, Bn, L4);
        glu16_k<<<EW(512 * P1d)>>>(fa, fb, h16, 512 * P1d);
        run_conv(w2, PAR(P_RES_B2) + i * 256, h16, fb, w16,
                 512, 1, L4, Bn, 1, 3, 1, 1, 0, 1, 256, 1, L4, stream);
        instnorm_k<<<256 * Bn, 256, 0, stream>>>(fb, PAR(P_RES_G2) + i * 256, PAR(P_RES_BE2) + i * 256, Bn, L4);
        add_k<<<EW(256 * P1d)>>>(fc, fb, 256 * P1d);
    }

    // ---- c12 1x1 (256->5120) ; RN_B ; reshape to (B,256,20,128) fp16 ----
    cvt16_k<<<EW(256 * P1d)>>>(fc, h16, 256 * P1d);
    run_conv(PAR(P_C12_W), PAR(P_C12_B), h16, fa, w16,
             256, 1, L4, Bn, 1, 1, 1, 1, 0, 0, 5120, 1, L4, stream);
    rnb_stats_k<<<5120, 256, 0, stream>>>(fa, mask, st, Bn, L4);
    rnb_apply_k<<<EW(5120 * P1d)>>>(fa, mask, st,
        PAR(P_RN1_FG), PAR(P_RN1_FB), PAR(P_RN1_BG), PAR(P_RN1_BB), fb, 5120, Bn, L4);
    reshape_1to2_h16_k<<<EW(256 * P2)>>>(fb, h16, 256, Bn, 20, L4);

    // ---- up1: conv 256->1024 5x5, pixel shuffle, IN, self-GLU ----
    run_conv(PAR(P_UP1_W), PAR(P_UP1_B), h16, fa, w16,
             256, 20, 128, Bn, 5, 5, 1, 1, 2, 2, 1024, 20, 128, stream);
    pshuf_k<<<EW(256 * P1)>>>(fa, fb, 256, Bn, 20, 128);
    instnorm_k<<<256 * Bn, 256, 0, stream>>>(fb, PAR(P_UP1_G), PAR(P_UP1_BE), Bn, 40 * 256);
    selfglu16_k<<<EW(256 * P1)>>>(fb, h16, 256 * P1);

    // ---- up2: conv 256->512 5x5, pixel shuffle, IN, self-GLU ----
    run_conv(PAR(P_UP2_W), PAR(P_UP2_B), h16, fa, w16,
             256, 40, 256, Bn, 5, 5, 1, 1, 2, 2, 512, 40, 256, stream);
    pshuf_k<<<EW(128 * P0)>>>(fa, fc, 128, Bn, 40, 256);
    instnorm_k<<<128 * Bn, 256, 0, stream>>>(fc, PAR(P_UP2_G), PAR(P_UP2_BE), Bn, 80 * 512);
    selfglu16_k<<<EW(128 * P0)>>>(fc, h16, 128 * P0);

    // ---- last conv 128->1, 5x15 -> writes (B,80,512) directly to d_out ----
    run_conv(PAR(P_LAST_W), PAR(P_LAST_B), h16, (float*)d_out, w16,
             128, 80, 512, Bn, 5, 15, 1, 1, 2, 7, 1, 80, 512, stream);

    #undef EW
    #undef PAR
}
